// Core_70660801953815
// MI455X (gfx1250) — matrix operand order repaired and checked by reading, not compiled, not run on hardware
//
#include <hip/hip_runtime.h>
#include <hip/hip_bf16.h>
#include <math.h>

// ---------------------------------------------------------------------------
// MI455X (gfx1250) implementation: bf16 WMMA GEMMs (fp32 weights read once
// from HBM, converted in-register with v_perm packing, fp32 accumulate) +
// fused gate/LN and attention kernels.  wave32 everywhere.
// ---------------------------------------------------------------------------

typedef __attribute__((ext_vector_type(16))) __bf16 v16bf;
typedef __attribute__((ext_vector_type(8)))  float  v8f;

union V16 { v16bf v; unsigned d[8]; };

// Two fp32 -> packed bf16x2 dword.  Round-half-up (u + 0x8000) then one
// v_perm_b32 extracts both high halves: 3 VALU per 2 elements.
__device__ __forceinline__ unsigned pack_bf16x2(float lo, float hi) {
    unsigned a = __builtin_bit_cast(unsigned, lo) + 0x8000u;
    unsigned b = __builtin_bit_cast(unsigned, hi) + 0x8000u;
    // result = { b[31:16], a[31:16] }
    return __builtin_amdgcn_perm(b, a, 0x07060302u);
}

// 8 consecutive fp32 (two float4) -> 4 packed dwords (8 bf16 elements)
__device__ __forceinline__ void packf8(unsigned* d, float4 a, float4 b) {
    d[0] = pack_bf16x2(a.x, a.y);
    d[1] = pack_bf16x2(a.z, a.w);
    d[2] = pack_bf16x2(b.x, b.y);
    d[3] = pack_bf16x2(b.z, b.w);
}

__device__ __forceinline__ float sigmoidf(float x) {
    return 1.0f / (1.0f + expf(-x));
}

// ---------------------------------------------------------------------------
// GEMM:  C[M,N] = A[M,K] * W[N,K]^T + bias[N]
// block = 128 threads (4 waves, 2x2);  block tile = 128 rows x 128 cols;
// wave tile = 64x64 = 4 M-tiles x 4 N-tiles; K loop step 32.
// 16 WMMAs per k-step, 128 converted elements/lane per k-step.
// Requires: M % 128 == 0, K % 32 == 0.  N arbitrary (clamped loads, masked
// stores).
// A fragment layout (16-bit A 16x32, same contiguous K order as B):
//   lane<16 : row=lane,    elements 0..15 = K k0+0..15
//   lane>=16: row=lane-16, elements 0..15 = K k0+16..31
// B fragment layout (K x 16):
//   lane<16 : col=lane,    elements 0..15 = K k0+0..15
//   lane>=16: col=lane-16, elements 0..15 = K k0+16..31
// ---------------------------------------------------------------------------
__global__ __launch_bounds__(128) void gemm_bf16w_kernel(
    const float* __restrict__ A, const float* __restrict__ W,
    const float* __restrict__ bias, float* __restrict__ C,
    int M, int N, int K)
{
    const int lane  = threadIdx.x & 31;
    const int wv    = threadIdx.x >> 5;
    const int lmod  = lane & 15;
    const int lhalf = lane >> 4;
    const int wr    = wv >> 1;
    const int wc    = wv & 1;

    const int mbase = blockIdx.y * 128 + wr * 64;
    const int nbase = blockIdx.x * 128 + wc * 64;

    int ncol[4];
    const float* wrow[4];
#pragma unroll
    for (int nt = 0; nt < 4; ++nt) {
        ncol[nt] = nbase + nt * 16 + lmod;
        const int nld = (ncol[nt] < N) ? ncol[nt] : (N - 1);
        wrow[nt] = W + (size_t)nld * K + lhalf * 16;
    }
    const float* arowb = A + (size_t)(mbase + lmod) * K + lhalf * 16;

    v8f acc[4][4];
#pragma unroll
    for (int i = 0; i < 4; ++i)
#pragma unroll
        for (int j = 0; j < 4; ++j)
#pragma unroll
            for (int r = 0; r < 8; ++r) acc[i][j][r] = 0.0f;

    for (int k0 = 0; k0 < K; k0 += 32) {
        // --- 4 B fragments: 16 contiguous fp32 along K of each column ---
        V16 bfr[4];
#pragma unroll
        for (int nt = 0; nt < 4; ++nt) {
            const float4* wp = (const float4*)(wrow[nt] + k0);
            float4 w0 = wp[0], w1 = wp[1], w2 = wp[2], w3 = wp[3];
            packf8(bfr[nt].d, w0, w1);
            packf8(bfr[nt].d + 4, w2, w3);
        }
        // --- 4 A fragments, 4 WMMAs each ---
#pragma unroll
        for (int mt = 0; mt < 4; ++mt) {
            const float* ar = arowb + (size_t)(mt * 16) * K + k0;
            float4 a0 = *(const float4*)(ar);
            float4 a1 = *(const float4*)(ar + 4);
            float4 a2 = *(const float4*)(ar + 8);
            float4 a3 = *(const float4*)(ar + 12);
            V16 afr;
            packf8(afr.d, a0, a1);
            packf8(afr.d + 4, a2, a3);
#pragma unroll
            for (int nt = 0; nt < 4; ++nt) {
                acc[mt][nt] = __builtin_amdgcn_wmma_f32_16x16x32_bf16(
                    false, afr.v, false, bfr[nt].v, (short)0, acc[mt][nt],
                    false, false);
            }
        }
    }

#pragma unroll
    for (int nt = 0; nt < 4; ++nt) {
        if (ncol[nt] < N) {
            const float bv = bias[ncol[nt]];
#pragma unroll
            for (int mt = 0; mt < 4; ++mt) {
#pragma unroll
                for (int r = 0; r < 8; ++r) {
                    const int row = mbase + mt * 16 + lhalf * 8 + r;
                    C[(size_t)row * N + ncol[nt]] = acc[mt][nt][r] + bv;
                }
            }
        }
    }
}

// ---------------------------------------------------------------------------
// gru_in = [relu(embed[it]), mean_fc, mean_att, sman_out]   [B,4096]
// ---------------------------------------------------------------------------
__global__ void build_gru_in_kernel(const int* __restrict__ it,
                                    const float* __restrict__ embed,
                                    const float* __restrict__ mfc,
                                    const float* __restrict__ mat,
                                    const float* __restrict__ sman,
                                    float* __restrict__ out)
{
    const int b = blockIdx.x, tid = threadIdx.x;
    const int row = it[b];
    float* o = out + (size_t)b * 4096;
    const float* e = embed + (size_t)row * 1024;
    for (int c = tid; c < 1024; c += 256) {
        o[c]        = fmaxf(e[c], 0.0f);
        o[1024 + c] = mfc[(size_t)b * 1024 + c];
        o[2048 + c] = mat[(size_t)b * 1024 + c];
        o[3072 + c] = sman[(size_t)b * 1024 + c];
    }
}

// ---------------------------------------------------------------------------
// Fused GRU gates + LayerNorm.  gi,gh: [B,3072]; hprev: [B,1024] -> out [B,1024]
// ---------------------------------------------------------------------------
__global__ void gru_ln_kernel(const float* __restrict__ gi,
                              const float* __restrict__ gh,
                              const float* __restrict__ hprev,
                              const float* __restrict__ g,
                              const float* __restrict__ be,
                              float* __restrict__ out)
{
    const int R = 1024;
    const int b = blockIdx.x, tid = threadIdx.x;
    __shared__ float s1[256], s2[256];
    __shared__ float hbuf[1024];

    const float* gib = gi + (size_t)b * 3072;
    const float* ghb = gh + (size_t)b * 3072;
    const float* hb  = hprev + (size_t)b * R;

    float a1 = 0.0f, a2 = 0.0f;
    for (int c = tid; c < R; c += 256) {
        float r = sigmoidf(gib[c] + ghb[c]);
        float z = sigmoidf(gib[c + 1024] + ghb[c + 1024]);
        float n = tanhf(gib[c + 2048] + r * ghb[c + 2048]);
        float h = (1.0f - z) * n + z * hb[c];
        hbuf[c] = h;
        a1 += h;
        a2 += h * h;
    }
    s1[tid] = a1; s2[tid] = a2;
    __syncthreads();
    for (int o = 128; o > 0; o >>= 1) {
        if (tid < o) { s1[tid] += s1[tid + o]; s2[tid] += s2[tid + o]; }
        __syncthreads();
    }
    const float mu  = s1[0] / R;
    const float var = s2[0] / R - mu * mu;
    const float inv = rsqrtf(var + 1e-5f);
    for (int c = tid; c < R; c += 256) {
        float h = hbuf[c];
        out[(size_t)b * R + c] = (h - mu) * inv * g[c] + be[c];
    }
}

// ---------------------------------------------------------------------------
// Fused attention: he [B,512], pfeat [B,F,512], Wa[512], ba[1], feats [B,F,1024]
// out[b,:] = softmax_f( sum_a tanh(pfeat+he)*Wa + ba ) . feats
// block = 256 threads (8 waves), one batch row per block.
// ---------------------------------------------------------------------------
__global__ void attention_kernel(const float* __restrict__ he,
                                 const float* __restrict__ pfeat,
                                 const float* __restrict__ Wa,
                                 const float* __restrict__ ba,
                                 const float* __restrict__ feats,
                                 float* __restrict__ out, int F)
{
    const int b = blockIdx.x;
    const int tid = threadIdx.x;
    const int lane = tid & 31;
    const int w = tid >> 5;
    __shared__ float sc[32];
    __shared__ float pi[32];

    const float* heb = he + (size_t)b * 512;
    for (int f = w; f < F; f += 8) {
        const float* pf = pfeat + ((size_t)b * F + f) * 512;
        float p = 0.0f;
        for (int a = lane; a < 512; a += 32)
            p += tanhf(pf[a] + heb[a]) * Wa[a];
        for (int off = 16; off > 0; off >>= 1)
            p += __shfl_xor(p, off);
        if (lane == 0) sc[f] = p + ba[0];
    }
    __syncthreads();

    float mx = -1e30f;
    for (int f = 0; f < F; ++f) mx = fmaxf(mx, sc[f]);
    float den = 0.0f;
    for (int f = 0; f < F; ++f) den += expf(sc[f] - mx);
    const float invd = 1.0f / den;
    if (tid < F) pi[tid] = expf(sc[tid] - mx) * invd;
    __syncthreads();

    for (int c = tid; c < 1024; c += 256) {
        float acc = 0.0f;
        for (int f = 0; f < F; ++f)
            acc += pi[f] * feats[((size_t)b * F + f) * 1024 + c];
        out[(size_t)b * 1024 + c] = acc;
    }
}

// ---------------------------------------------------------------------------
// Concats
// ---------------------------------------------------------------------------
__global__ void concat3_kernel(const float* __restrict__ x0,
                               const float* __restrict__ x1,
                               const float* __restrict__ x2,
                               float* __restrict__ out)
{
    const int b = blockIdx.x, tid = threadIdx.x;
    for (int c = tid; c < 1024; c += 256) {
        out[(size_t)b * 3072 + c]        = x0[(size_t)b * 1024 + c];
        out[(size_t)b * 3072 + 1024 + c] = x1[(size_t)b * 1024 + c];
        out[(size_t)b * 3072 + 2048 + c] = x2[(size_t)b * 1024 + c];
    }
}

__global__ void concat4_kernel(const float* __restrict__ x0,
                               const float* __restrict__ x1,
                               const float* __restrict__ x2,
                               const float* __restrict__ x3,
                               float* __restrict__ out)
{
    const int b = blockIdx.x, tid = threadIdx.x;
    for (int c = tid; c < 1024; c += 256) {
        out[(size_t)b * 4096 + c]        = x0[(size_t)b * 1024 + c];
        out[(size_t)b * 4096 + 1024 + c] = x1[(size_t)b * 1024 + c];
        out[(size_t)b * 4096 + 2048 + c] = x2[(size_t)b * 1024 + c];
        out[(size_t)b * 4096 + 3072 + c] = x3[(size_t)b * 1024 + c];
    }
}

// out[b,f,:] = feats[b,f,:] for f<T, = extra[b,:] for f==T    (grid: (T+1, B))
__global__ void catfeat_kernel(const float* __restrict__ feats,
                               const float* __restrict__ extra,
                               float* __restrict__ out, int T)
{
    const int f = blockIdx.x, b = blockIdx.y, tid = threadIdx.x;
    const float* src = (f < T) ? (feats + ((size_t)b * T + f) * 1024)
                               : (extra + (size_t)b * 1024);
    float* dst = out + ((size_t)b * (T + 1) + f) * 1024;
    for (int c = tid; c < 1024; c += 256) dst[c] = src[c];
}

// tail: hn, attn, smann, hm, hv, hn  (6 x 131072 floats)  grid (512, 6)
__global__ void pack_tail_kernel(const float* __restrict__ hn,
                                 const float* __restrict__ attn,
                                 const float* __restrict__ smann,
                                 const float* __restrict__ hm,
                                 const float* __restrict__ hv,
                                 float* __restrict__ dst)
{
    const int seg = blockIdx.y;
    const size_t i = (size_t)blockIdx.x * blockDim.x + threadIdx.x;
    if (i >= 131072) return;
    const float* s;
    switch (seg) {
        case 0: s = hn; break;
        case 1: s = attn; break;
        case 2: s = smann; break;
        case 3: s = hm; break;
        case 4: s = hv; break;
        default: s = hn; break;
    }
    dst[(size_t)seg * 131072 + i] = s[i];
}

// ---------------------------------------------------------------------------
// Host orchestration
// ---------------------------------------------------------------------------
extern "C" void kernel_launch(void* const* d_in, const int* in_sizes, int n_in,
                              void* d_out, int out_size, void* d_ws, size_t ws_size,
                              hipStream_t stream)
{
    (void)in_sizes; (void)n_in; (void)out_size; (void)ws_size;

    const int*   it        = (const int*)  d_in[0];
    const float* mean_fc   = (const float*)d_in[1];
    const float* fc_feats  = (const float*)d_in[2];
    const float* p_fc      = (const float*)d_in[3];
    const float* mean_att  = (const float*)d_in[4];
    const float* att_feats = (const float*)d_in[5];
    const float* p_att     = (const float*)d_in[6];
    const float* state     = (const float*)d_in[7];
    const float* att_out   = (const float*)d_in[8];
    const float* sman_out  = (const float*)d_in[9];
    const float* motion_f  = (const float*)d_in[10];
    const float* visual_f  = (const float*)d_in[11];
    const float* text_f    = (const float*)d_in[12];
    const float* embed     = (const float*)d_in[13];
    const float* gru_Wih   = (const float*)d_in[14];
    const float* gru_Whh   = (const float*)d_in[15];
    const float* gru_bih   = (const float*)d_in[16];
    const float* gru_bhh   = (const float*)d_in[17];
    const float* gru_ng    = (const float*)d_in[18];
    const float* gru_nb    = (const float*)d_in[19];
    const float* ag_Wih    = (const float*)d_in[20];
    const float* ag_Whh    = (const float*)d_in[21];
    const float* ag_bih    = (const float*)d_in[22];
    const float* ag_bhh    = (const float*)d_in[23];
    const float* att_ng    = (const float*)d_in[24];
    const float* att_nb    = (const float*)d_in[25];
    const float* sg_Wih    = (const float*)d_in[26];
    const float* sg_Whh    = (const float*)d_in[27];
    const float* sg_bih    = (const float*)d_in[28];
    const float* sg_bhh    = (const float*)d_in[29];
    const float* sman_ng   = (const float*)d_in[30];
    const float* sman_nb   = (const float*)d_in[31];
    const float* v2a_W = (const float*)d_in[32]; const float* v2a_b = (const float*)d_in[33];
    const float* m2a_W = (const float*)d_in[34]; const float* m2a_b = (const float*)d_in[35];
    const float* t2a_W = (const float*)d_in[36]; const float* t2a_b = (const float*)d_in[37];
    const float* l1_W  = (const float*)d_in[38]; const float* l1_b  = (const float*)d_in[39];
    const float* l2_W  = (const float*)d_in[40]; const float* l2_b  = (const float*)d_in[41];
    const float* l3_W  = (const float*)d_in[42]; const float* l3_b  = (const float*)d_in[43];
    const float* mot_Wh  = (const float*)d_in[44]; const float* mot_bh  = (const float*)d_in[45];
    const float* mot_Wa  = (const float*)d_in[46]; const float* mot_ba  = (const float*)d_in[47];
    const float* vis_Wh  = (const float*)d_in[48]; const float* vis_bh  = (const float*)d_in[49];
    const float* vis_Wa  = (const float*)d_in[50]; const float* vis_ba  = (const float*)d_in[51];
    const float* cmot_Wh = (const float*)d_in[52]; const float* cmot_bh = (const float*)d_in[53];
    const float* cmot_Wa = (const float*)d_in[54]; const float* cmot_ba = (const float*)d_in[55];
    const float* cvis_Wh = (const float*)d_in[56]; const float* cvis_bh = (const float*)d_in[57];
    const float* cvis_Wa = (const float*)d_in[58]; const float* cvis_ba = (const float*)d_in[59];
    const float* ctxt_Wh = (const float*)d_in[60]; const float* ctxt_bh = (const float*)d_in[61];
    const float* ctxt_Wa = (const float*)d_in[62]; const float* ctxt_ba = (const float*)d_in[63];

    float* ws  = (float*)d_ws;
    float* out = (float*)d_out;

    // workspace layout (floats); slots reused across dead stages
    float* X     = ws;                  // 524288: gru_in -> att_in -> sman_in
    float* GI    = ws + 524288;         // 393216: gi -> agi -> sgi
    float* GH    = ws + 917504;         // 393216: gh -> agh -> sgh
    float* HN    = ws + 1310720;        // 131072
    float* ATTN  = ws + 1441792;        // 131072
    float* SMANN = ws + 1572864;        // 131072
    float* HMOT  = ws + 1703936;        // 131072
    float* HVIS  = ws + 1835008;        // 131072
    float* HE    = ws + 1966080;        // 65536 (reused by all 5 attentions)
    float* CM    = ws + 2031616;        // 131072
    float* CV    = ws + 2162688;        // 131072
    float* CT    = ws + 2293760;        // 131072
    float* CF    = ws + 2424832;        // 2752512 (mf/vf/tf, sequential)
    float* PF    = ws + 5177344;        // 1376256 (p_mf/p_vf/p_tf, sequential)
    // total: 6,553,600 floats = 25 MiB

    const int B = 128;
    dim3 gblk(128);
    auto gemm = [&](const float* A, const float* W, const float* bias, float* C,
                    int M, int N, int K) {
        dim3 grid((N + 127) / 128, M / 128);
        gemm_bf16w_kernel<<<grid, gblk, 0, stream>>>(A, W, bias, C, M, N, K);
    };

    // -------- language GRU --------
    build_gru_in_kernel<<<B, 256, 0, stream>>>(it, embed, mean_fc, mean_att, sman_out, X);
    gemm(X,     gru_Wih, gru_bih, GI, B, 3072, 4096);
    gemm(state, gru_Whh, gru_bhh, GH, B, 3072, 1024);
    gru_ln_kernel<<<B, 256, 0, stream>>>(GI, GH, state, gru_ng, gru_nb, HN);

    // -------- motion / visual attention --------
    gemm(HN, mot_Wh, mot_bh, HE, B, 512, 1024);
    attention_kernel<<<B, 256, 0, stream>>>(HE, p_fc, mot_Wa, mot_ba, fc_feats, HMOT, 30);
    gemm(HN, vis_Wh, vis_bh, HE, B, 512, 1024);
    attention_kernel<<<B, 256, 0, stream>>>(HE, p_att, vis_Wa, vis_ba, att_feats, HVIS, 30);

    // -------- attention GRU --------
    concat3_kernel<<<B, 256, 0, stream>>>(HMOT, HVIS, HN, X);
    gemm(X,       ag_Wih, ag_bih, GI, B, 3072, 3072);
    gemm(att_out, ag_Whh, ag_bhh, GH, B, 3072, 1024);
    gru_ln_kernel<<<B, 256, 0, stream>>>(GI, GH, att_out, att_ng, att_nb, ATTN);

    // -------- motion/visual/text context (sequential, slot-reusing) --------
    struct Mod {
        const float* feats; const float* extra;
        const float* pW; const float* pb;
        const float* Wh; const float* bh; const float* Wa; const float* ba;
        float* dst;
    };
    const Mod mods[3] = {
        { motion_f, HMOT, m2a_W, m2a_b, cmot_Wh, cmot_bh, cmot_Wa, cmot_ba, CM },
        { visual_f, HVIS, v2a_W, v2a_b, cvis_Wh, cvis_bh, cvis_Wa, cvis_ba, CV },
        { text_f,   HN,   t2a_W, t2a_b, ctxt_Wh, ctxt_bh, ctxt_Wa, ctxt_ba, CT },
    };
    for (int i = 0; i < 3; ++i) {
        const Mod& m = mods[i];
        dim3 cgrid(21, B);
        catfeat_kernel<<<cgrid, 256, 0, stream>>>(m.feats, m.extra, CF, 20);
        gemm(CF, m.pW, m.pb, PF, B * 21, 512, 1024);
        gemm(ATTN, m.Wh, m.bh, HE, B, 512, 1024);
        attention_kernel<<<B, 256, 0, stream>>>(HE, PF, m.Wa, m.ba, CF, m.dst, 21);
    }

    // -------- SMAN GRU --------
    concat4_kernel<<<B, 256, 0, stream>>>(CM, CV, CT, ATTN, X);
    gemm(X,        sg_Wih, sg_bih, GI, B, 3072, 4096);
    gemm(sman_out, sg_Whh, sg_bhh, GH, B, 3072, 1024);
    gru_ln_kernel<<<B, 256, 0, stream>>>(GI, GH, sman_out, sman_ng, sman_nb, SMANN);

    // -------- logits (write directly into d_out) --------
    const int V = 20001;
    gemm(HN,    l1_W, l1_b, out,                     B, V, 1024);
    gemm(ATTN,  l2_W, l2_b, out + (size_t)B * V,     B, V, 1024);
    gemm(SMANN, l3_W, l3_b, out + (size_t)2 * B * V, B, V, 1024);

    // -------- tail outputs: h_n, att_n, sman_n, h_motion, h_visual, h_n --------
    dim3 tgrid(512, 6);
    pack_tail_kernel<<<tgrid, 256, 0, stream>>>(HN, ATTN, SMANN, HMOT, HVIS,
                                                out + (size_t)3 * B * V);
}
